// VarianceAdaptor_26130581029018
// MI455X (gfx1250) — matrix operand order repaired and checked by reading, not compiled, not run on hardware
//
#include <hip/hip_runtime.h>
#include <stdint.h>

typedef __attribute__((ext_vector_type(16))) _Float16 v16h;
typedef __attribute__((ext_vector_type(8)))  _Float16 v8h;
typedef __attribute__((ext_vector_type(8)))  float    v8f;

#define Bn   64
#define Sn   256
#define Tn   1024
#define Hn   256
#define Fn   256
#define Kw   3
#define Kdim 768          // Kw*Hn, kk = k*Hn + h
#define Mtot (Bn*Sn)      // 16384
#define NBn  256
#define MT   32           // M rows per block (2 WMMA subtiles)

// ---------------------------------------------------------------------------
// Weight prep: w[f][h][k] (f32) -> Wt[kk][f] (f16), kk = k*256 + h, 6 layers.
// ---------------------------------------------------------------------------
__global__ void wprep_kernel(const float* __restrict__ w0, const float* __restrict__ w1,
                             const float* __restrict__ w2, const float* __restrict__ w3,
                             const float* __restrict__ w4, const float* __restrict__ w5,
                             _Float16* __restrict__ wt) {
  int gid   = blockIdx.x * blockDim.x + threadIdx.x;  // 6*768*256 total
  int layer = gid / (Kdim * Fn);
  int rem   = gid % (Kdim * Fn);
  int kk    = rem / Fn;
  int f     = rem % Fn;
  int k = kk >> 8, h = kk & 255;
  const float* src = layer == 0 ? w0 : layer == 1 ? w1 : layer == 2 ? w2
                   : layer == 3 ? w3 : layer == 4 ? w4 : w5;
  wt[gid] = (_Float16)src[(f * Hn + h) * Kw + k];
}

// ---------------------------------------------------------------------------
// Fused conv1d(K=3,SAME) -> +bias -> ReLU -> LayerNorm, via WMMA GEMM.
// Block = 256 threads (8 waves). Block tile: 32 rows x 256 cols, K = 768.
// Wave w owns N cols [32w,32w+32) and both 16-row M subtiles: each B fragment
// loaded from L2 feeds 2 WMMAs (halves L2 weight traffic vs 16-row tiles).
// ---------------------------------------------------------------------------
__global__ void __launch_bounds__(256) convln_kernel(
    const float* __restrict__ Xin, const _Float16* __restrict__ Wt,
    const float* __restrict__ bias, const float* __restrict__ gamma,
    const float* __restrict__ beta, float* __restrict__ Out) {
  // A tile (32x768 f16 = 48 KB) aliased with epilogue tile (32x256 f32 = 32 KB):
  // A is dead after the K loop; a barrier separates the lifetimes.
  __shared__ __align__(16) char smem[MT * Kdim * 2];
  _Float16* Alds = (_Float16*)smem;
  float*    Clds = (float*)smem;
  __shared__ float red[2][MT][8];
  __shared__ float stat[2][MT];

  const int tid = threadIdx.x;
  const int m0  = blockIdx.x * MT;

  // im2col load: Alds[row][k*256+h] = Xin[b, s+k-1, h] (0 at sequence edges)
  for (int idx = tid; idx < MT * Kdim; idx += 256) {
    int row = idx / Kdim;
    int kk  = idx % Kdim;
    int k = kk >> 8, h = kk & 255;
    int m  = m0 + row;
    int b  = m >> 8;
    int s  = m & 255;
    int sp = s + k - 1;
    float v = 0.f;
    if (sp >= 0 && sp < Sn) v = Xin[(((size_t)b << 8) + sp) * Hn + h];
    Alds[row * Kdim + kk] = (_Float16)v;
  }
  __syncthreads();

  const int wave  = tid >> 5;
  const int lane  = tid & 31;
  const int row16 = lane & 15;
  const int half  = lane >> 4;
  const int ncol0 = wave * 32;
  const int krow_lane = row16 + (half << 4);   // B-matrix K row owned by lane

  v8f acc00 = {};   // msub 0, nsub 0
  v8f acc01 = {};   // msub 0, nsub 1
  v8f acc10 = {};   // msub 1, nsub 0
  v8f acc11 = {};   // msub 1, nsub 1

  for (int k0 = 0; k0 < Kdim; k0 += 32) {
    // A fragments for both 16-row subtiles:
    // elems 0..7 = K k0+8*half.., elems 8..15 = K k0+16+8*half..
    const int abase = k0 + (half << 3);
    const v8h a0lo = *(const v8h*)&Alds[row16 * Kdim + abase];
    const v8h a0hi = *(const v8h*)&Alds[row16 * Kdim + abase + 16];
    const v8h a1lo = *(const v8h*)&Alds[(row16 + 16) * Kdim + abase];
    const v8h a1hi = *(const v8h*)&Alds[(row16 + 16) * Kdim + abase + 16];
    v16h a0, a1;
#pragma unroll
    for (int i = 0; i < 8; ++i) {
      a0[i] = a0lo[i]; a0[i + 8] = a0hi[i];
      a1[i] = a1lo[i]; a1[i + 8] = a1hi[i];
    }

    // B fragments: lane = N col (lane&15), elems in same K order as A
    const _Float16* brow = Wt + (size_t)(k0 + krow_lane) * Fn + ncol0;
    const _Float16* bcol = Wt + (size_t)(k0 + (half << 3)) * Fn + ncol0 + row16;
    v16h b0, b1;
#pragma unroll
    for (int i = 0; i < 8; ++i) {
      b0[i] = bcol[i * Fn];        b0[i + 8] = bcol[(16 + i) * Fn];
      b1[i] = bcol[i * Fn + 16];   b1[i + 8] = bcol[(16 + i) * Fn + 16];
    }
    if (k0 + 32 < Kdim)
      __builtin_prefetch(brow + 32 * Fn, 0, 1);   // global_prefetch_b8

    acc00 = __builtin_amdgcn_wmma_f32_16x16x32_f16(
        false, a0, false, b0, (short)0, acc00, false, false);
    acc01 = __builtin_amdgcn_wmma_f32_16x16x32_f16(
        false, a0, false, b1, (short)0, acc01, false, false);
    acc10 = __builtin_amdgcn_wmma_f32_16x16x32_f16(
        false, a1, false, b0, (short)0, acc10, false, false);
    acc11 = __builtin_amdgcn_wmma_f32_16x16x32_f16(
        false, a1, false, b1, (short)0, acc11, false, false);
  }
  __syncthreads();   // A tile dead; smem now reused as the epilogue C tile

  // Scatter C tiles (N = lane&15, M = r + 8*half [+16 for subtile 1]) + bias/ReLU
  {
    const int ccol = lane & 15;
    const float bb0 = bias[ncol0 + ccol];
    const float bb1 = bias[ncol0 + 16 + ccol];
#pragma unroll
    for (int r = 0; r < 8; ++r) {
      int crow0 = r + (half << 3);
      int crow1 = crow0 + 16;
      float v00 = acc00[r] + bb0, v01 = acc01[r] + bb1;
      float v10 = acc10[r] + bb0, v11 = acc11[r] + bb1;
      Clds[crow0 * Fn + ncol0 + ccol]      = v00 > 0.f ? v00 : 0.f;
      Clds[crow0 * Fn + ncol0 + 16 + ccol] = v01 > 0.f ? v01 : 0.f;
      Clds[crow1 * Fn + ncol0 + ccol]      = v10 > 0.f ? v10 : 0.f;
      Clds[crow1 * Fn + ncol0 + 16 + ccol] = v11 > 0.f ? v11 : 0.f;
    }
  }
  __syncthreads();

  // LayerNorm across Fn: 8 threads per row, 32 elems per thread, 32 rows
  const int rrow = tid >> 3;
  const int rcol = tid & 7;
  float s = 0.f, sq = 0.f;
#pragma unroll
  for (int j = 0; j < 32; ++j) {
    float v = Clds[rrow * Fn + rcol + j * 8];
    s += v; sq += v * v;
  }
  red[0][rrow][rcol] = s;
  red[1][rrow][rcol] = sq;
  __syncthreads();
  if (rcol == 0) {
    float ts = 0.f, tq = 0.f;
#pragma unroll
    for (int j = 0; j < 8; ++j) { ts += red[0][rrow][j]; tq += red[1][rrow][j]; }
    float mean = ts * (1.f / Fn);
    float var  = tq * (1.f / Fn) - mean * mean;
    stat[0][rrow] = mean;
    stat[1][rrow] = rsqrtf(var + 1e-5f);
  }
  __syncthreads();
  const float mean = stat[0][rrow];
  const float rstd = stat[1][rrow];
#pragma unroll
  for (int j = 0; j < 32; ++j) {
    int col = rcol + j * 8;
    float v = (Clds[rrow * Fn + col] - mean) * rstd * gamma[col] + beta[col];
    Out[(size_t)(m0 + rrow) * Fn + col] = v;
  }
}

// ---------------------------------------------------------------------------
// Final projection: out[m] = mask ? 0 : dot(H[m], wl) + bl. One wave per row.
// ---------------------------------------------------------------------------
__global__ void pred_kernel(const float* __restrict__ Hin, const float* __restrict__ wl,
                            const float* __restrict__ bl,
                            const unsigned char* __restrict__ mask,
                            float* __restrict__ out) {
  const int lane = threadIdx.x & 31;
  const int wave = threadIdx.x >> 5;
  const int m    = blockIdx.x * 8 + wave;
  const float* h = Hin + (size_t)m * Fn;
  float s = 0.f;
#pragma unroll
  for (int j = 0; j < 8; ++j) {
    int c = lane + j * 32;
    s += h[c] * wl[c];
  }
#pragma unroll
  for (int off = 16; off > 0; off >>= 1) s += __shfl_down(s, off, 32);
  if (lane == 0) out[m] = mask[m] ? 0.f : (s + bl[0]);
}

// ---------------------------------------------------------------------------
// searchsorted(bins, tgt, 'left') for pitch (first 16K ids) and energy (next)
// ---------------------------------------------------------------------------
__global__ void bins_kernel(const float* __restrict__ tp, const float* __restrict__ te,
                            const float* __restrict__ binsP, const float* __restrict__ binsE,
                            int* __restrict__ idxP, int* __restrict__ idxE) {
  int gid   = blockIdx.x * 256 + threadIdx.x;     // 2*Mtot total
  int which = gid >> 14;
  int i     = gid & (Mtot - 1);
  const float* bins = which ? binsE : binsP;
  float v = which ? te[i] : tp[i];
  int lo = 0, hi = NBn - 1;                       // 255 boundaries
  while (lo < hi) {
    int mid = (lo + hi) >> 1;
    if (bins[mid] < v) lo = mid + 1; else hi = mid;
  }
  (which ? idxE : idxP)[i] = lo;
}

// ---------------------------------------------------------------------------
// Per-batch inclusive cumsum of durations (Hillis-Steele) + duration copy-out
// ---------------------------------------------------------------------------
__global__ void cumsum_kernel(const int* __restrict__ dur, int* __restrict__ cum,
                              int* __restrict__ total, float* __restrict__ durOut) {
  __shared__ int buf[2][Sn];
  const int b = blockIdx.x, t = threadIdx.x;
  int v = dur[b * Sn + t];
  durOut[b * Sn + t] = (float)v;
  buf[0][t] = v;
  __syncthreads();
  int src = 0;
  for (int off = 1; off < Sn; off <<= 1) {
    int val = buf[src][t] + (t >= off ? buf[src][t - off] : 0);
    buf[src ^ 1][t] = val;
    __syncthreads();
    src ^= 1;
  }
  cum[b * Sn + t] = buf[src][t];
  if (t == Sn - 1) total[b] = buf[src][t];
}

// ---------------------------------------------------------------------------
// Length regulate fused with h = x + emb_pitch[idxP] + emb_energy[idxE].
// Grid (T, B), 256 threads = one output channel each.
// ---------------------------------------------------------------------------
__global__ void regulate_kernel(const float* __restrict__ x,
                                const float* __restrict__ embP, const float* __restrict__ embE,
                                const int* __restrict__ idxP, const int* __restrict__ idxE,
                                const int* __restrict__ cum, const int* __restrict__ total,
                                float* __restrict__ out, float* __restrict__ maskOut) {
  const int t = blockIdx.x;
  const int b = blockIdx.y;
  const int c = threadIdx.x;
  const int* cb = cum + b * Sn;
  int lo = 0, hi = Sn;                            // searchsorted 'right'
  while (lo < hi) {
    int mid = (lo + hi) >> 1;
    if (cb[mid] <= t) lo = mid + 1; else hi = mid;
  }
  int i = lo > Sn - 1 ? Sn - 1 : lo;
  int tot = total[b];
  int lim = tot < Tn ? tot : Tn;
  bool valid = t < lim;
  int src = b * Sn + i;
  float v = x[(size_t)src * Hn + c]
          + embP[(size_t)idxP[src] * Hn + c]
          + embE[(size_t)idxE[src] * Hn + c];
  out[((size_t)(b * Tn + t)) * Hn + c] = valid ? v : 0.f;
  if (c == 0) maskOut[b * Tn + t] = valid ? 0.f : 1.f;
}

// ---------------------------------------------------------------------------
extern "C" void kernel_launch(void* const* d_in, const int* in_sizes, int n_in,
                              void* d_out, int out_size, void* d_ws, size_t ws_size,
                              hipStream_t stream) {
  const float*         x            = (const float*)d_in[0];
  const unsigned char* src_mask     = (const unsigned char*)d_in[1];
  const float*         tgt_pitch    = (const float*)d_in[2];
  const float*         tgt_energy   = (const float*)d_in[3];
  const int*           tgt_duration = (const int*)d_in[4];
  // d_in[5] = tgt_max_length (compile-time Tn)

  // params_{dur,pitch,energy}: w1,b1,g1,be1,w2,b2,g2,be2,wl,bl
  const float* P[3][10];
  for (int p = 0; p < 3; ++p)
    for (int j = 0; j < 10; ++j)
      P[p][j] = (const float*)d_in[6 + p * 10 + j];
  const float* emb_pitch   = (const float*)d_in[36];
  const float* emb_energy  = (const float*)d_in[37];
  const float* bins_pitch  = (const float*)d_in[38];
  const float* bins_energy = (const float*)d_in[39];

  // workspace layout
  char* ws = (char*)d_ws;
  _Float16* Wt   = (_Float16*)ws;                               // 6*768*256 f16 = 2,359,296 B
  float*    hA   = (float*)(ws + 2359296);                      // 16 MB
  float*    hB   = (float*)(ws + 2359296 + 16777216);           // 16 MB
  int*      idxP = (int*)(ws + 2359296 + 2 * 16777216);
  int*      idxE = idxP + Mtot;
  int*      cum  = idxE + Mtot;
  int*      total = cum + Mtot;

  // d_out layout: out(B,T,H) | p_pred | e_pred | dur_pred | tgt_duration | tgt_mask
  float* out0     = (float*)d_out;
  float* p_pred   = out0 + (size_t)Bn * Tn * Hn;
  float* e_pred   = p_pred + Mtot;
  float* dur_pred = e_pred + Mtot;
  float* dur_copy = dur_pred + Mtot;
  float* mask_out = dur_copy + Mtot;

  // 1) transpose/convert all six conv weights to f16
  wprep_kernel<<<6 * Kdim * Fn / 256, 256, 0, stream>>>(
      P[0][0], P[0][4], P[1][0], P[1][4], P[2][0], P[2][4], Wt);

  // 2) three predictors: conv+LN, conv+LN, projection
  const size_t WtL = (size_t)Kdim * Fn;
  float* preds[3] = { dur_pred, p_pred, e_pred };
  for (int p = 0; p < 3; ++p) {
    convln_kernel<<<Mtot / MT, 256, 0, stream>>>(
        x, Wt + (size_t)(2 * p) * WtL, P[p][1], P[p][2], P[p][3], hA);
    convln_kernel<<<Mtot / MT, 256, 0, stream>>>(
        hA, Wt + (size_t)(2 * p + 1) * WtL, P[p][5], P[p][6], P[p][7], hB);
    pred_kernel<<<Mtot / 8, 256, 0, stream>>>(hB, P[p][8], P[p][9], src_mask, preds[p]);
  }

  // 3) embedding bin indices, duration cumsum, fused length-regulate gather
  bins_kernel<<<2 * Mtot / 256, 256, 0, stream>>>(
      tgt_pitch, tgt_energy, bins_pitch, bins_energy, idxP, idxE);
  cumsum_kernel<<<Bn, Sn, 0, stream>>>(tgt_duration, cum, total, dur_copy);
  dim3 grid(Tn, Bn);
  regulate_kernel<<<grid, Hn, 0, stream>>>(
      x, emb_pitch, emb_energy, idxP, idxE, cum, total, out0, mask_out);
}
